// GNN_26594437497717
// MI455X (gfx1250) — matrix operand order repaired and checked by reading, not compiled, not run on hardware
//
#include <hip/hip_runtime.h>
#include <hip/hip_bf16.h>
#include <math.h>

typedef __attribute__((ext_vector_type(16))) _Float16 v16h;
typedef __attribute__((ext_vector_type(8)))  float    v8f;

#define HID 128
#define OUTD 7
#define GRAPHS 64
#define EPSV 1e-5f
#define SLOPE 0.2f

// ---------------- utility ----------------

__global__ void k_fill(float* __restrict__ p, float v, long n) {
  long i = (long)blockIdx.x * blockDim.x + threadIdx.x;
  if (i < n) p[i] = v;
}

__device__ __forceinline__ void atomicMaxF(float* addr, float val) {
  // works with slots initialized to -inf
  if (val >= 0.0f) atomicMax((int*)addr, __float_as_int(val));
  else             atomicMin((unsigned int*)addr, __float_as_uint(val));
}

// ---------------- layer-1 feature transform: x(N x 4) @ W(4 x 128) ----------------

__global__ void k_transform1(const float* __restrict__ x,
                             const float* __restrict__ Wl,
                             const float* __restrict__ Wr,
                             float* __restrict__ XL, float* __restrict__ XR, int N) {
  long idx = (long)blockIdx.x * blockDim.x + threadIdx.x;
  if (idx >= (long)N * HID) return;
  int n = (int)(idx >> 7);
  int j = (int)(idx & 127);
  float4 xv = *(const float4*)(x + (long)n * 4);
  float sl = xv.x * Wl[j] + xv.y * Wl[HID + j] + xv.z * Wl[2 * HID + j] + xv.w * Wl[3 * HID + j];
  float sr = xv.x * Wr[j] + xv.y * Wr[HID + j] + xv.z * Wr[2 * HID + j] + xv.w * Wr[3 * HID + j];
  XL[idx] = sl;
  XR[idx] = sr;
}

// ---------------- WMMA GEMM: H(nrows x 128) @ W(128 x 128) -> Out(nrows x 128) ----------------
// fp32 -> fp16 inputs, fp32 accumulate via v_wmma_f32_16x16x32_f16.
// 256-thread block = 8 waves; each wave owns a 16-row x 128-col output stripe.

__global__ void __launch_bounds__(256)
k_gemm128_wmma(const float* __restrict__ Hm, const float* __restrict__ W,
               float* __restrict__ Out, int nrows) {
  __shared__ __align__(32) _Float16 sBt[128 * 128]; // transposed weights: [col][k], 32KB
  int tid = threadIdx.x;
  #pragma unroll 4
  for (int idx = tid; idx < 128 * 128; idx += 256) {
    int j = idx >> 7, k = idx & 127;
    sBt[idx] = (_Float16)W[k * 128 + j];
  }
  __syncthreads();

  int wave = tid >> 5, lane = tid & 31;
  int rowBase = blockIdx.x * 128 + wave * 16;
  // wave-uniform: the whole 16-row tile is in range (nrows % 16 == 0 makes
  // partial tiles impossible for this workload; generic fallback kept anyway)
  bool tileFull = (rowBase + 16) <= nrows;

  // --- A fragments: 16x32 fp16 tiles, 4 K-chunks, contiguous k order (k = 16h + s), matched by B ---
  // lane<16 holds row=lane, K = kc*32+0..15; lane>=16 same row, K = kc*32+16..31
  int m = lane & 15;
  int row = rowBase + m;
  int kb = (lane < 16) ? 0 : 16;
  const float* hr = Hm + (long)row * 128;

  v16h a[4];
  if (tileFull) {
    #pragma unroll
    for (int kc = 0; kc < 4; ++kc) {
      int base = kc * 32 + kb;
      float4 p0 = *(const float4*)(hr + base);
      float4 p1 = *(const float4*)(hr + base + 4);
      float4 p2 = *(const float4*)(hr + base + 8);
      float4 p3 = *(const float4*)(hr + base + 12);
      a[kc][0]  = (_Float16)p0.x; a[kc][1]  = (_Float16)p0.y;
      a[kc][2]  = (_Float16)p0.z; a[kc][3]  = (_Float16)p0.w;
      a[kc][4]  = (_Float16)p1.x; a[kc][5]  = (_Float16)p1.y;
      a[kc][6]  = (_Float16)p1.z; a[kc][7]  = (_Float16)p1.w;
      a[kc][8]  = (_Float16)p2.x; a[kc][9]  = (_Float16)p2.y;
      a[kc][10] = (_Float16)p2.z; a[kc][11] = (_Float16)p2.w;
      a[kc][12] = (_Float16)p3.x; a[kc][13] = (_Float16)p3.y;
      a[kc][14] = (_Float16)p3.z; a[kc][15] = (_Float16)p3.w;
    }
  } else {
    bool valid = row < nrows;
    #pragma unroll
    for (int kc = 0; kc < 4; ++kc) {
      #pragma unroll
      for (int i = 0; i < 16; ++i) a[kc][i] = (_Float16)0.0f;
      if (valid) {
        int base = kc * 32 + kb;
        #pragma unroll
        for (int i = 0; i < 8; ++i) a[kc][i] = (_Float16)hr[base + i];
        #pragma unroll
        for (int i = 0; i < 8; ++i) a[kc][8 + i] = (_Float16)hr[base + 8 + i];
      }
    }
  }

  // --- B fragments from LDS (transposed layout -> contiguous 32B per lane) ---
  // B 32x16: lane<16 holds K=0..15 of column (lane&15); lane>=16 holds K=16..31
  int ncol = lane & 15;
  int kbB = (lane < 16) ? 0 : 16;
  int mstore0 = rowBase + ((lane < 16) ? 0 : 8);

  #pragma unroll
  for (int ct = 0; ct < 8; ++ct) {
    int col = ct * 16 + ncol;
    const _Float16* bp = sBt + col * 128 + kbB;
    v16h b0 = *(const v16h*)(bp);
    v16h b1 = *(const v16h*)(bp + 32);
    v16h b2 = *(const v16h*)(bp + 64);
    v16h b3 = *(const v16h*)(bp + 96);
    v8f c = {};
    c = __builtin_amdgcn_wmma_f32_16x16x32_f16(false, a[0], false, b0, (short)0, c, false, false);
    c = __builtin_amdgcn_wmma_f32_16x16x32_f16(false, a[1], false, b1, (short)0, c, false, false);
    c = __builtin_amdgcn_wmma_f32_16x16x32_f16(false, a[2], false, b2, (short)0, c, false, false);
    c = __builtin_amdgcn_wmma_f32_16x16x32_f16(false, a[3], false, b3, (short)0, c, false, false);

    // C/D layout: VGPR r -> (M = r + (lane<16?0:8), N = lane&15)
    float* op = Out + (long)mstore0 * 128 + col;
    if (tileFull) {
      #pragma unroll
      for (int r = 0; r < 8; ++r) op[(long)r * 128] = c[r];
    } else {
      #pragma unroll
      for (int r = 0; r < 8; ++r)
        if (mstore0 + r < nrows) op[(long)r * 128] = c[r];
    }
  }
}

// ---------------- edge attention logits + segment max (one wave per edge) ----------------

__global__ void k_edge_logits(const float* __restrict__ XL, const float* __restrict__ XR,
                              const float* __restrict__ att, const int* __restrict__ ei,
                              float* __restrict__ P, float* __restrict__ EMAX,
                              int E, int EP) {
  int wave = threadIdx.x >> 5;
  int lane = threadIdx.x & 31;
  long e = (long)blockIdx.x * 8 + wave;
  if (e >= EP) return;
  int src, dst;
  if (e < E) { src = ei[e]; dst = ei[(long)E + e]; }
  else       { src = dst = (int)(e - E); }
  int c0 = lane * 4;  // lanes 0..15 -> head 0 (ch 0..63), lanes 16..31 -> head 1
  float4 xl4 = *(const float4*)(XL + (long)src * HID + c0);
  float4 xr4 = *(const float4*)(XR + (long)dst * HID + c0);
  float4 a4  = *(const float4*)(att + c0);
  float mval, acc = 0.0f;
  mval = xl4.x + xr4.x; acc += (mval > 0.0f ? mval : SLOPE * mval) * a4.x;
  mval = xl4.y + xr4.y; acc += (mval > 0.0f ? mval : SLOPE * mval) * a4.y;
  mval = xl4.z + xr4.z; acc += (mval > 0.0f ? mval : SLOPE * mval) * a4.z;
  mval = xl4.w + xr4.w; acc += (mval > 0.0f ? mval : SLOPE * mval) * a4.w;
  // reduce within each 16-lane half (per head)
  acc += __shfl_xor(acc, 8, 16);
  acc += __shfl_xor(acc, 4, 16);
  acc += __shfl_xor(acc, 2, 16);
  acc += __shfl_xor(acc, 1, 16);
  if ((lane & 15) == 0) {
    int h = lane >> 4;
    P[e * 2 + h] = acc;
    atomicMaxF(&EMAX[(long)dst * 2 + h], acc);
  }
}

// ---------------- softmax numerator + segment denominator ----------------

__global__ void k_edge_softmax(const int* __restrict__ ei, float* __restrict__ P,
                               const float* __restrict__ EMAX, float* __restrict__ DEN,
                               int E, int EP) {
  long idx = (long)blockIdx.x * blockDim.x + threadIdx.x;
  if (idx >= (long)EP * 2) return;
  long e = idx >> 1;
  int h = (int)(idx & 1);
  int dst = (e < E) ? ei[(long)E + e] : (int)(e - E);
  float p = __expf(P[idx] - EMAX[(long)dst * 2 + h]);
  P[idx] = p;
  atomicAdd(&DEN[(long)dst * 2 + h], p);
}

// ---------------- weighted aggregation (one wave per edge, float4 lanes) ----------------

__global__ void k_edge_agg(const float* __restrict__ XL, const float* __restrict__ P,
                           const float* __restrict__ DEN, const int* __restrict__ ei,
                           float* __restrict__ ACC, int E, int EP) {
  int wave = threadIdx.x >> 5;
  int lane = threadIdx.x & 31;
  long e = (long)blockIdx.x * 8 + wave;
  if (e >= EP) return;
  int src, dst;
  if (e < E) { src = ei[e]; dst = ei[(long)E + e]; }
  else       { src = dst = (int)(e - E); }
  int h = lane >> 4;
  float alpha = P[e * 2 + h] / DEN[(long)dst * 2 + h];
  int c0 = lane * 4;
  float4 x4 = *(const float4*)(XL + (long)src * HID + c0);
  float* o = ACC + (long)dst * HID + c0;
  atomicAdd(o + 0, alpha * x4.x);
  atomicAdd(o + 1, alpha * x4.y);
  atomicAdd(o + 2, alpha * x4.z);
  atomicAdd(o + 3, alpha * x4.w);
}

// ---------------- batchnorm stats (per-channel sum / sumsq of conv_out + bias) ----------------

__global__ void k_bn_stats(const float* __restrict__ ACC, const float* __restrict__ bias,
                           float* __restrict__ SUM, float* __restrict__ SSQ, int N) {
  int j = threadIdx.x;          // 128 threads = channels
  int r0 = blockIdx.x * 64;     // 64 rows per block
  float b = bias[j], s = 0.0f, ss = 0.0f;
  for (int r = 0; r < 64; ++r) {
    int row = r0 + r;
    if (row < N) {
      float v = ACC[(long)row * HID + j] + b;
      s += v; ss += v * v;
    }
  }
  atomicAdd(&SUM[j], s);
  atomicAdd(&SSQ[j], ss);
}

// ---------------- batchnorm apply + ReLU (in place) ----------------

__global__ void k_bn_apply(float* __restrict__ ACC, const float* __restrict__ bias,
                           const float* __restrict__ gam, const float* __restrict__ bet,
                           const float* __restrict__ SUM, const float* __restrict__ SSQ, int N) {
  long idx = (long)blockIdx.x * blockDim.x + threadIdx.x;
  if (idx >= (long)N * HID) return;
  int j = (int)(idx & 127);
  float invN = 1.0f / (float)N;
  float mu = SUM[j] * invN;
  float var = SSQ[j] * invN - mu * mu;
  float v = ACC[idx] + bias[j];
  float y = gam[j] * (v - mu) * rsqrtf(var + EPSV) + bet[j];
  ACC[idx] = fmaxf(y, 0.0f);
}

// ---------------- global mean pool (sums + counts) ----------------

__global__ void k_pool(const float* __restrict__ ACC, const int* __restrict__ batch,
                       float* __restrict__ POOL, float* __restrict__ CNT, int N) {
  long idx = (long)blockIdx.x * blockDim.x + threadIdx.x;
  if (idx >= (long)N * HID) return;
  int n = (int)(idx >> 7);
  int j = (int)(idx & 127);
  int g = batch[n];
  atomicAdd(&POOL[(long)g * HID + j], ACC[idx]);
  if (j == 0) atomicAdd(&CNT[g], 1.0f);
}

// ---------------- MLP head: relu(pooled @ W3 + b3) @ W4 + b4 ----------------

__global__ void k_head(const float* __restrict__ POOL, const float* __restrict__ CNT,
                       const float* __restrict__ W3, const float* __restrict__ b3,
                       const float* __restrict__ W4, const float* __restrict__ b4,
                       float* __restrict__ out) {
  __shared__ float sT[GRAPHS * 64];
  int tid = threadIdx.x;  // 256
  for (int t = tid; t < GRAPHS * 64; t += 256) {
    int g = t >> 6, j = t & 63;
    float cnt = fmaxf(CNT[g], 1.0f);
    float s = b3[j];
    for (int k = 0; k < HID; ++k)
      s += (POOL[(long)g * HID + k] / cnt) * W3[k * 64 + j];
    sT[t] = fmaxf(s, 0.0f);
  }
  __syncthreads();
  for (int t = tid; t < GRAPHS * OUTD; t += 256) {
    int g = t / OUTD, o = t % OUTD;
    float s = b4[o];
    for (int k = 0; k < 64; ++k)
      s += sT[g * 64 + k] * W4[k * OUTD + o];
    out[t] = s;
  }
}

// ---------------- launcher ----------------

extern "C" void kernel_launch(void* const* d_in, const int* in_sizes, int n_in,
                              void* d_out, int out_size, void* d_ws, size_t ws_size,
                              hipStream_t stream) {
  const float* x    = (const float*)d_in[0];
  const int*   ei   = (const int*)d_in[1];
  const int*   batch= (const int*)d_in[2];
  const float* Wl1  = (const float*)d_in[3];
  const float* Wr1  = (const float*)d_in[4];
  const float* att1 = (const float*)d_in[5];
  const float* b1   = (const float*)d_in[6];
  const float* g1   = (const float*)d_in[7];
  const float* be1  = (const float*)d_in[8];
  const float* Wl2  = (const float*)d_in[9];
  const float* Wr2  = (const float*)d_in[10];
  const float* att2 = (const float*)d_in[11];
  const float* b2   = (const float*)d_in[12];
  const float* g2   = (const float*)d_in[13];
  const float* be2  = (const float*)d_in[14];
  const float* W3   = (const float*)d_in[15];
  const float* b3   = (const float*)d_in[16];
  const float* W4   = (const float*)d_in[17];
  const float* b4   = (const float*)d_in[18];
  float* out = (float*)d_out;

  const int N  = in_sizes[0] / 4;   // 50000
  const int E  = in_sizes[1] / 2;   // 800000
  const int EP = E + N;             // edges + self-loops

  // workspace layout (floats)
  float* ws   = (float*)d_ws;
  float* XL   = ws;                       // N*128
  float* XR   = XL + (long)N * HID;       // N*128
  float* ACC  = XR + (long)N * HID;       // N*128 (conv out -> h -> next conv out)
  float* P    = ACC + (long)N * HID;      // EP*2 (logits then softmax numerators)
  float* EMAX = P + (long)EP * 2;         // N*2
  float* DEN  = EMAX + (long)N * 2;       // N*2
  float* SUM  = DEN + (long)N * 2;        // 128
  float* SSQ  = SUM + HID;                // 128
  float* POOL = SSQ + HID;                // 64*128
  float* CNT  = POOL + (long)GRAPHS * HID;// 64

  const int TPB = 256;
  dim3 blk(TPB);
  long nElem = (long)N * HID;
  dim3 gElem((unsigned)((nElem + TPB - 1) / TPB));
  dim3 gEdgeWave((unsigned)((EP + 7) / 8));             // one wave32 per edge
  dim3 gEdge2((unsigned)(((long)EP * 2 + TPB - 1) / TPB));
  dim3 gGemm((unsigned)((N + 127) / 128));
  dim3 gStats((unsigned)((N + 63) / 64));

  #define FILL(p, v, n) k_fill<<<dim3((unsigned)(((n) + TPB - 1) / TPB)), blk, 0, stream>>>((p), (v), (long)(n))

  // ======== layer 1 ========
  FILL(ACC, 0.0f, nElem);
  FILL(EMAX, -INFINITY, (long)N * 2);
  FILL(DEN, 0.0f, (long)N * 2);
  FILL(SUM, 0.0f, 2 * HID);  // SUM + SSQ contiguous

  k_transform1<<<gElem, blk, 0, stream>>>(x, Wl1, Wr1, XL, XR, N);
  k_edge_logits<<<gEdgeWave, blk, 0, stream>>>(XL, XR, att1, ei, P, EMAX, E, EP);
  k_edge_softmax<<<gEdge2, blk, 0, stream>>>(ei, P, EMAX, DEN, E, EP);
  k_edge_agg<<<gEdgeWave, blk, 0, stream>>>(XL, P, DEN, ei, ACC, E, EP);
  k_bn_stats<<<gStats, dim3(HID), 0, stream>>>(ACC, b1, SUM, SSQ, N);
  k_bn_apply<<<gElem, blk, 0, stream>>>(ACC, b1, g1, be1, SUM, SSQ, N);  // ACC = h1

  // ======== layer 2 ========
  k_gemm128_wmma<<<gGemm, blk, 0, stream>>>(ACC, Wl2, XL, N);
  k_gemm128_wmma<<<gGemm, blk, 0, stream>>>(ACC, Wr2, XR, N);

  FILL(ACC, 0.0f, nElem);
  FILL(EMAX, -INFINITY, (long)N * 2);
  FILL(DEN, 0.0f, (long)N * 2);
  FILL(SUM, 0.0f, 2 * HID);

  k_edge_logits<<<gEdgeWave, blk, 0, stream>>>(XL, XR, att2, ei, P, EMAX, E, EP);
  k_edge_softmax<<<gEdge2, blk, 0, stream>>>(ei, P, EMAX, DEN, E, EP);
  k_edge_agg<<<gEdgeWave, blk, 0, stream>>>(XL, P, DEN, ei, ACC, E, EP);
  k_bn_stats<<<gStats, dim3(HID), 0, stream>>>(ACC, b2, SUM, SSQ, N);
  k_bn_apply<<<gElem, blk, 0, stream>>>(ACC, b2, g2, be2, SUM, SSQ, N);  // ACC = h2

  // ======== pool + head ========
  FILL(POOL, 0.0f, (long)GRAPHS * HID + GRAPHS);  // POOL + CNT contiguous
  k_pool<<<gElem, blk, 0, stream>>>(ACC, batch, POOL, CNT, N);
  k_head<<<dim3(1), blk, 0, stream>>>(POOL, CNT, W3, b3, W4, b4, out);

  #undef FILL
}
